// LurieNet_5617817223341
// MI455X (gfx1250) — matrix operand order repaired and checked by reading, not compiled, not run on hardware
//
#include <hip/hip_runtime.h>
#include <cstddef>

// LurieNet recurrence on gfx1250 (MI455X).
// Each block owns a 16-row batch tile and iterates all 255 time steps locally
// (rows are independent across time). GEMMs use v_wmma_f32_16x16x32_f16 with
// f32 accumulation; carried state X stays f32 in LDS; weights converted once
// to row-major f16 in workspace (B-fragment friendly layout).

namespace {
constexpr int   kN    = 512;   // state dim
constexpr int   kBS   = 512;   // batch
constexpr int   kT    = 256;   // time steps (output length)
constexpr float kStep = 0.05f;
}

typedef __attribute__((ext_vector_type(16))) _Float16 v16h;
typedef __attribute__((ext_vector_type(8)))  _Float16 v8h;
typedef __attribute__((ext_vector_type(8)))  float    v8f;

__device__ __forceinline__ v16h cat8(v8h lo, v8h hi) {
  return __builtin_shufflevector(lo, hi, 0,1,2,3,4,5,6,7,8,9,10,11,12,13,14,15);
}

__device__ __forceinline__ float fast_tanh(float x) {
  // tanh(x) = 1 - 2/(exp(2x)+1); exact limits at +/-inf, one v_exp_f32.
  float e = __expf(2.0f * x);
  return 1.0f - 2.0f / (e + 1.0f);
}

// ---- one-shot f32 -> f16 weight conversion into workspace: [A | B | C] ----
__global__ void cvt_weights_kernel(const float* __restrict__ A,
                                   const float* __restrict__ B,
                                   const float* __restrict__ C,
                                   _Float16* __restrict__ w) {
  int i = blockIdx.x * blockDim.x + threadIdx.x;
  constexpr int M = kN * kN;
  if (i >= 3 * M) return;
  float v;
  if (i < M)          v = A[i];
  else if (i < 2 * M) v = B[i - M];
  else                v = C[i - 2 * M];
  w[i] = (_Float16)v;
}

// ---- main recurrent kernel: 1 block = 16 batch rows, 4 waves split N ----
__launch_bounds__(128, 1)
__global__ void lurie_kernel(const float* __restrict__ X0,
                             const _Float16* __restrict__ Wa,
                             const _Float16* __restrict__ Wb,
                             const _Float16* __restrict__ Wc,
                             const float* __restrict__ bx,
                             const float* __restrict__ by,
                             float* __restrict__ out) {
  __shared__ float    Xs[16 * kN];   // 32 KB: f32 master state tile
  __shared__ _Float16 Xh[16 * kN];   // 16 KB: f16 copy (WMMA A-fragments)
  __shared__ _Float16 Sh[16 * kN];   // 16 KB: tanh activations (f16)

  const int tid  = threadIdx.x;
  const int lane = tid & 31;
  const int wave = tid >> 5;         // 0..3, owns cols [wave*128, +128)
  const int l16  = lane & 15;
  const int lh   = lane >> 4;        // 0/1: which half-wave
  const int row0 = blockIdx.x * 16;  // batch row base

  // ---- load X0 tile into LDS; emit t=0 output ----
  for (int i = tid; i < 16 * kN; i += 128) {
    int m = i >> 9;            // i / 512
    int n = i & (kN - 1);
    float v = X0[(size_t)(row0 + m) * kN + n];
    Xs[i] = v;
    Xh[i] = (_Float16)v;
    out[((size_t)(row0 + m) * kT + 0) * kN + n] = v;
  }
  __syncthreads();

  // A-fragment source pointers for this lane (this code's k order:
  // lanes 0-15 hold K 0..15, lanes 16-31 hold K 16..31).
  const _Float16* xrow = &Xh[l16 * kN + lh * 16];
  const _Float16* srow = &Sh[l16 * kN + lh * 16];

  for (int t = 1; t < kT; ++t) {
    // ================= phase 1: S = tanh(X @ C^T + by) =================
    for (int j = 0; j < 8; ++j) {
      const int ncol = (wave * 8 + j) * 16 + l16;
      // B-fragment: 16 contiguous f16 of row ncol of C (lanes 0-15: K 0..15,
      // lanes 16-31: K 16..31) -> row-major f16 weights need no repack.
      const _Float16* wrow = Wc + (size_t)ncol * kN + lh * 16;
      v8f acc = {};
      for (int k = 0; k < kN; k += 32) {
        v16h a = cat8(*(const v8h*)(xrow + k), *(const v8h*)(xrow + k + 8));
        v16h b = cat8(*(const v8h*)(wrow + k), *(const v8h*)(wrow + k + 8));
        acc = __builtin_amdgcn_wmma_f32_16x16x32_f16(
            false, a, false, b, (short)0, acc, false, false);
      }
      const float byv = by[ncol];
#pragma unroll
      for (int v = 0; v < 8; ++v) {
        int m = v + lh * 8;                       // C/D layout: M = v + 8*lh
        Sh[m * kN + ncol] = (_Float16)fast_tanh(acc[v] + byv);
      }
    }
    __syncthreads();

    // ====== phase 2: Xn = X + step*(X @ A^T + S @ B^T + bx) -> out ======
    for (int j = 0; j < 8; ++j) {
      const int ncol = (wave * 8 + j) * 16 + l16;
      const _Float16* arow = Wa + (size_t)ncol * kN + lh * 16;
      const _Float16* brow = Wb + (size_t)ncol * kN + lh * 16;
      v8f acc = {};
      for (int k = 0; k < kN; k += 32) {
        v16h ax = cat8(*(const v8h*)(xrow + k), *(const v8h*)(xrow + k + 8));
        v16h ba = cat8(*(const v8h*)(arow + k), *(const v8h*)(arow + k + 8));
        acc = __builtin_amdgcn_wmma_f32_16x16x32_f16(
            false, ax, false, ba, (short)0, acc, false, false);
        v16h as = cat8(*(const v8h*)(srow + k), *(const v8h*)(srow + k + 8));
        v16h bb = cat8(*(const v8h*)(brow + k), *(const v8h*)(brow + k + 8));
        acc = __builtin_amdgcn_wmma_f32_16x16x32_f16(
            false, as, false, bb, (short)0, acc, false, false);
      }
      const float bxv = bx[ncol];
#pragma unroll
      for (int v = 0; v < 8; ++v) {
        int m = v + lh * 8;
        float xold = Xs[m * kN + ncol];
        float x = xold + kStep * (acc[v] + bxv);
        out[((size_t)(row0 + m) * kT + t) * kN + ncol] = x;
      }
    }
    __syncthreads();   // all waves done reading X/Xh/Sh for this step

    // ===== phase 3: commit new X from just-written (L2-hot) output =====
    for (int j = 0; j < 8; ++j) {
      const int ncol = (wave * 8 + j) * 16 + l16;
#pragma unroll
      for (int v = 0; v < 8; ++v) {
        int m = v + lh * 8;
        float x = out[((size_t)(row0 + m) * kT + t) * kN + ncol];
        Xs[m * kN + ncol] = x;
        Xh[m * kN + ncol] = (_Float16)x;
      }
    }
    __syncthreads();
  }
}

extern "C" void kernel_launch(void* const* d_in, const int* in_sizes, int n_in,
                              void* d_out, int out_size, void* d_ws, size_t ws_size,
                              hipStream_t stream) {
  (void)in_sizes; (void)n_in; (void)out_size; (void)ws_size;
  const float* X0 = (const float*)d_in[0];
  const float* A  = (const float*)d_in[1];
  const float* B  = (const float*)d_in[2];
  const float* C  = (const float*)d_in[3];
  const float* bx = (const float*)d_in[4];
  const float* by = (const float*)d_in[5];
  float* out = (float*)d_out;

  _Float16* w = (_Float16*)d_ws;            // 3 * 512*512 f16 = 1.5 MB scratch
  constexpr int M = kN * kN;

  cvt_weights_kernel<<<(3 * M + 255) / 256, 256, 0, stream>>>(A, B, C, w);
  lurie_kernel<<<kBS / 16, 128, 0, stream>>>(X0, w, w + M, w + 2 * M,
                                             bx, by, out);
}
